// _KNN_58763742544405
// MI455X (gfx1250) — matrix operand order repaired and checked by reading, not compiled, not run on hardware
//
#include <hip/hip_runtime.h>
#include <hip/hip_bf16.h>
#include <math.h>

// ---------------- problem constants (match reference) ----------------
#define BQ      1024        // queries
#define DIM     256         // feature dim (K of the GEMM)
#define NTRAIN  200000      // train rows (N of the GEMM)
#define NCLASS  1000
#define KTOP    20
#define TEMP    20.0f
#define EPS     1e-8f

#define NCHUNK  2048                          // train rows per block (128 tiles of 16)
#define NCHUNKS ((NTRAIN + NCHUNK - 1) / NCHUNK)   // 98
#define NCAND   (NCHUNKS * KTOP)              // 1960 candidates per query

typedef __attribute__((ext_vector_type(16))) _Float16 v16h;
typedef __attribute__((ext_vector_type(8)))  _Float16 v8h;
typedef __attribute__((ext_vector_type(8)))  float    v8f;

// ---------------- prep: standardize + L2-normalize queries, split hi/lo f16 ----
__global__ __launch_bounds__(256)
void knn_prep_query(const float* __restrict__ x, const float* __restrict__ mean,
                    const float* __restrict__ inv_std,
                    _Float16* __restrict__ qh, _Float16* __restrict__ ql) {
    __shared__ float red[DIM];
    const int b = blockIdx.x, d = threadIdx.x;
    float v = (x[(size_t)b * DIM + d] - mean[d]) * inv_std[d];
    red[d] = v * v;
    __syncthreads();
    for (int s = DIM / 2; s > 0; s >>= 1) {
        if (d < s) red[d] += red[d + s];
        __syncthreads();
    }
    const float inv = 1.0f / fmaxf(sqrtf(red[0]), EPS);
    const float qv = v * inv;
    const _Float16 hi = (_Float16)qv;
    qh[(size_t)b * DIM + d] = hi;
    ql[(size_t)b * DIM + d] = (_Float16)(qv - (float)hi);
}

// ---------------- prep: L2-normalize train bank, split hi/lo f16 --------------
__global__ __launch_bounds__(256)
void knn_prep_train(const float* __restrict__ tx,
                    _Float16* __restrict__ th, _Float16* __restrict__ tl) {
    __shared__ float red[DIM];
    const int n = blockIdx.x, d = threadIdx.x;
    float v = tx[(size_t)n * DIM + d];
    red[d] = v * v;
    __syncthreads();
    for (int s = DIM / 2; s > 0; s >>= 1) {
        if (d < s) red[d] += red[d + s];
        __syncthreads();
    }
    const float inv = 1.0f / fmaxf(sqrtf(red[0]), EPS);
    const float qv = v * inv;
    const _Float16 hi = (_Float16)qv;
    th[(size_t)n * DIM + d] = hi;
    tl[(size_t)n * DIM + d] = (_Float16)(qv - (float)hi);
}

// A-fragment loader for v_wmma 16x16x32 f16 (this code's contiguous k order, B uses the same):
// lane (l&15)=row M; lanes 0-15 hold K=0..15 contiguous;
// lanes 16-31 hold K=16..31.  base already points at row*DIM + s*32 + 16*(l>>4).
__device__ inline v16h load_a_frag(const _Float16* __restrict__ base) {
    v8h lo = *(const v8h*)(base);
    v8h hi = *(const v8h*)(base + 8);
    v16h r;
#pragma unroll
    for (int i = 0; i < 8; ++i) { r[i] = lo[i]; r[i + 8] = hi[i]; }
    return r;
}

// ---------------- fused GEMM (3-term hi/lo f16 WMMA, f32 accum) + chunk top-20 --
// Three independent accumulator chains (one per precision term) so the 24 WMMAs
// per tile form 3 interleavable chains of 8 instead of one serial chain of 24.
__global__ __launch_bounds__(32)
void knn_gemm_topk(const _Float16* __restrict__ qh, const _Float16* __restrict__ ql,
                   const _Float16* __restrict__ th, const _Float16* __restrict__ tl,
                   float* __restrict__ cand_vals, int* __restrict__ cand_idx) {
    __shared__ float lds_sim[16 * 16];

    const int l      = threadIdx.x;       // 0..31, one wave32 per block
    const int lane16 = l & 15;
    const int half   = l >> 4;
    const int m0     = blockIdx.x * 16;   // query tile base
    const int chunk  = blockIdx.y;
    const int nbase  = chunk * NCHUNK;

    // Hoist the 16-query A fragments (hi and lo) into registers for the whole chunk.
    v16h ahi[8], alo[8];
#pragma unroll
    for (int s = 0; s < 8; ++s) {
        const size_t aoff = (size_t)(m0 + lane16) * DIM + s * 32 + 16 * half;
        ahi[s] = load_a_frag(qh + aoff);
        alo[s] = load_a_frag(ql + aoff);
    }

    float topv[KTOP];
    int   topi[KTOP];
#pragma unroll
    for (int j = 0; j < KTOP; ++j) { topv[j] = -3.0e38f; topi[j] = 0; }

    for (int t = 0; t < NCHUNK / 16; ++t) {
        const int tn = nbase + t * 16;
        if (tn >= NTRAIN) break;                       // wave-uniform
        int rowt = tn + lane16;
        if (rowt > NTRAIN - 1) rowt = NTRAIN - 1;      // clamp for partial tile
        // 32-bit element offsets (max ~51.2M elements < 2^31); pointers once per tile
        const unsigned roff = (unsigned)rowt * DIM + 16u * half;
        const _Float16* tb_h = th + roff;
        const _Float16* tb_l = tl + roff;

        v8f acc0 = {};   // hi * hi
        v8f acc1 = {};   // hi * lo
        v8f acc2 = {};   // lo * hi
#pragma unroll
        for (int s = 0; s < 8; ++s) {
            // B fragment: lane column = lane16, contiguous 16 halfs of K
            v16h bh = *(const v16h*)(tb_h + s * 32);
            v16h bl = *(const v16h*)(tb_l + s * 32);
            acc0 = __builtin_amdgcn_wmma_f32_16x16x32_f16(false, ahi[s], false, bh,
                                                          (short)0, acc0, false, false);
            acc1 = __builtin_amdgcn_wmma_f32_16x16x32_f16(false, ahi[s], false, bl,
                                                          (short)0, acc1, false, false);
            acc2 = __builtin_amdgcn_wmma_f32_16x16x32_f16(false, alo[s], false, bh,
                                                          (short)0, acc2, false, false);
        }

        // Transpose C fragment through LDS: lane holds col=lane16, rows r+8*half.
#pragma unroll
        for (int r = 0; r < 8; ++r)
            lds_sim[(r + 8 * half) * 16 + lane16] = acc0[r] + acc1[r] + acc2[r];
        __syncthreads();

        if (l < 16) {   // 16 lanes each own one query row; fail-fast top-20 insert
#pragma unroll
            for (int c = 0; c < 16; ++c) {
                const int gi = tn + c;
                const float v = lds_sim[l * 16 + c];
                if (gi < NTRAIN && v > topv[KTOP - 1]) {
                    float cv = v; int ci = gi;
#pragma unroll
                    for (int j = 0; j < KTOP; ++j) {
                        if (cv > topv[j]) {
                            float tv = topv[j]; topv[j] = cv; cv = tv;
                            int   ti = topi[j]; topi[j] = ci; ci = ti;
                        }
                    }
                }
            }
        }
        __syncthreads();
    }

    if (l < 16) {
        const size_t base = ((size_t)(m0 + l) * NCHUNKS + chunk) * KTOP;
#pragma unroll
        for (int j = 0; j < KTOP; ++j) {
            cand_vals[base + j] = topv[j];
            cand_idx[base + j]  = topi[j];
        }
    }
}

// ---------------- merge candidates -> global top-20 -> softmax -> class row ----
__global__ __launch_bounds__(32)
void knn_finalize(const float* __restrict__ cand_vals, const int* __restrict__ cand_idx,
                  const int* __restrict__ train_y, float* __restrict__ out) {
    __shared__ float sv[32 * KTOP];
    __shared__ int   si[32 * KTOP];
    __shared__ float srow[NCLASS];

    const int q = blockIdx.x, l = threadIdx.x;

    float tv[KTOP]; int ti[KTOP];
#pragma unroll
    for (int j = 0; j < KTOP; ++j) { tv[j] = -3.0e38f; ti[j] = 0; }

    for (int i = l; i < NCAND; i += 32) {
        const float v = cand_vals[(size_t)q * NCAND + i];
        if (v > tv[KTOP - 1]) {
            float cv = v; int ci = cand_idx[(size_t)q * NCAND + i];
#pragma unroll
            for (int j = 0; j < KTOP; ++j) {
                if (cv > tv[j]) {
                    float x = tv[j]; tv[j] = cv; cv = x;
                    int   y = ti[j]; ti[j] = ci; ci = y;
                }
            }
        }
    }
#pragma unroll
    for (int j = 0; j < KTOP; ++j) { sv[l * KTOP + j] = tv[j]; si[l * KTOP + j] = ti[j]; }
    for (int c = l; c < NCLASS; c += 32) srow[c] = 0.0f;
    __syncthreads();

    if (l == 0) {
        float fv[KTOP]; int fi[KTOP];
#pragma unroll
        for (int j = 0; j < KTOP; ++j) { fv[j] = -3.0e38f; fi[j] = 0; }
        for (int i = 0; i < 32 * KTOP; ++i) {
            const float v = sv[i];
            if (v > fv[KTOP - 1]) {
                float cv = v; int ci = si[i];
#pragma unroll
                for (int j = 0; j < KTOP; ++j) {
                    if (cv > fv[j]) {
                        float x = fv[j]; fv[j] = cv; cv = x;
                        int   y = fi[j]; fi[j] = ci; ci = y;
                    }
                }
            }
        }
        // softmax over fv*TEMP (fv sorted descending -> fv[0] is the max)
        float w[KTOP]; float sum = 0.0f;
#pragma unroll
        for (int j = 0; j < KTOP; ++j) { w[j] = __expf((fv[j] - fv[0]) * TEMP); sum += w[j]; }
        const float inv = 1.0f / sum;
#pragma unroll
        for (int j = 0; j < KTOP; ++j) srow[train_y[fi[j]]] += w[j] * inv;
    }
    __syncthreads();

    for (int c = l; c < NCLASS; c += 32) out[(size_t)q * NCLASS + c] = srow[c];
}

// ---------------- launcher ----------------------------------------------------
extern "C" void kernel_launch(void* const* d_in, const int* in_sizes, int n_in,
                              void* d_out, int out_size, void* d_ws, size_t ws_size,
                              hipStream_t stream) {
    const float* x       = (const float*)d_in[0];
    const float* mean    = (const float*)d_in[1];
    const float* inv_std = (const float*)d_in[2];
    const float* train_x = (const float*)d_in[3];
    const int*   train_y = (const int*)d_in[4];
    float*       out     = (float*)d_out;

    // workspace carve-up (all region sizes are multiples of 256 B)
    char* ws = (char*)d_ws;
    size_t off = 0;
    _Float16* qh = (_Float16*)(ws + off); off += (size_t)BQ * DIM * 2;       // 512 KB
    _Float16* ql = (_Float16*)(ws + off); off += (size_t)BQ * DIM * 2;       // 512 KB
    _Float16* th = (_Float16*)(ws + off); off += (size_t)NTRAIN * DIM * 2;   // 102.4 MB
    _Float16* tl = (_Float16*)(ws + off); off += (size_t)NTRAIN * DIM * 2;   // 102.4 MB
    float* cand_vals = (float*)(ws + off); off += (size_t)BQ * NCAND * 4;    // 8.0 MB
    int*   cand_idx  = (int*)(ws + off);   off += (size_t)BQ * NCAND * 4;    // 8.0 MB

    knn_prep_query<<<BQ, DIM, 0, stream>>>(x, mean, inv_std, qh, ql);
    knn_prep_train<<<NTRAIN, DIM, 0, stream>>>(train_x, th, tl);
    knn_gemm_topk<<<dim3(BQ / 16, NCHUNKS), 32, 0, stream>>>(qh, ql, th, tl,
                                                             cand_vals, cand_idx);
    knn_finalize<<<BQ, 32, 0, stream>>>(cand_vals, cand_idx, train_y, out);
}
